// EncoderLayer_75282186764563
// MI455X (gfx1250) — compile-verified, matrix operand order checked by reading, not run on hardware
//
#include <hip/hip_runtime.h>
#include <math.h>

// ---------------------------------------------------------------------------
// EncoderLayer for MI455X (gfx1250): 3x MHA + FFN.
// f16 operands + f32 accumulation via v_wmma_f32_16x16x32_f16 (wave32).
// Every WMMA fragment is fed by 2x global_load_b128 (f16 data laid out so the
// ISA 7.12.2 half-index pattern {k0+8*hi .. +7} ∪ {k0+16+8*hi .. +7} is two
// contiguous 16B runs). Softmax/LayerNorm fully vectorized to b128 accesses.
// ---------------------------------------------------------------------------

typedef __attribute__((ext_vector_type(16))) _Float16 v16h;
typedef __attribute__((ext_vector_type(8)))  _Float16 h8;
typedef __attribute__((ext_vector_type(4)))  _Float16 h4;
typedef __attribute__((ext_vector_type(8)))  float    v8f;
typedef __attribute__((ext_vector_type(4)))  float    f4;

#define S_LEN   1024
#define D_MODEL 512
#define N_HEAD  8
#define D_Kh    64
#define D_INNER 2048
#define M_TOT   (4 * S_LEN)        // 4096 rows (B*S)
#define RATIO_F 0.2f
#define TB      256                // 8 waves/block

#define WMMA_F16(a, b, c) \
  __builtin_amdgcn_wmma_f32_16x16x32_f16(false, (a), false, (b), (short)0, (c), false, false)

// f16 fragment: two contiguous 8-half (16B) runs -> 2x b128 loads
__device__ __forceinline__ v16h load_frag_h(const _Float16* __restrict__ p, int hi) {
  h8 lo = *(const h8*)(p + hi * 8);
  h8 hh = *(const h8*)(p + 16 + hi * 8);
  return __builtin_shufflevector(lo, hh, 0,1,2,3,4,5,6,7,8,9,10,11,12,13,14,15);
}

// f32 fragment: 4x b128 (float4) + packed cvt
__device__ __forceinline__ v16h load_frag_f(const float* __restrict__ p, int hi) {
  f4 a0 = *(const f4*)(p + hi * 8);
  f4 a1 = *(const f4*)(p + hi * 8 + 4);
  f4 b0 = *(const f4*)(p + 16 + hi * 8);
  f4 b1 = *(const f4*)(p + 16 + hi * 8 + 4);
  v16h r;
#pragma unroll
  for (int i = 0; i < 4; ++i) {
    r[i]      = (_Float16)a0[i];
    r[4 + i]  = (_Float16)a1[i];
    r[8 + i]  = (_Float16)b0[i];
    r[12 + i] = (_Float16)b1[i];
  }
  return r;
}

__device__ __forceinline__ float wave_sum(float x) {
#pragma unroll
  for (int m = 16; m >= 1; m >>= 1) x += __shfl_xor(x, m, 32);
  return x;
}
__device__ __forceinline__ float wave_max(float x) {
#pragma unroll
  for (int m = 16; m >= 1; m >>= 1) x = fmaxf(x, __shfl_xor(x, m, 32));
  return x;
}

// -------------------- f32 -> f16 flat convert (8 elems/thread) -------------
__global__ void cvt_f16(const float* __restrict__ src, _Float16* __restrict__ dst, int n8) {
  const int i = blockIdx.x * blockDim.x + threadIdx.x;
  if (i >= n8) return;
  f4 a = *(const f4*)(src + (size_t)i * 8);
  f4 b = *(const f4*)(src + (size_t)i * 8 + 4);
  h8 o;
#pragma unroll
  for (int e = 0; e < 4; ++e) { o[e] = (_Float16)a[e]; o[4 + e] = (_Float16)b[e]; }
  *(h8*)(dst + (size_t)i * 8) = o;
}

// -------------------- W[K,N] f32 -> Wt[N,K] f16 -----------------------------
__global__ void wt_cvt(const float* __restrict__ W, _Float16* __restrict__ Wt, int K, int N) {
  const int i = blockIdx.x * blockDim.x + threadIdx.x;
  if (i >= K * N) return;
  const int n = i / K, k = i - n * K;
  Wt[i] = (_Float16)W[(size_t)k * N + n];
}

// ---- GEMM: C[M,N] = A[M,K](f16,row) @ Wt[N,K](f16,row)ᵀ, wave = 16x64 ------
__global__ void gemm_h(const _Float16* __restrict__ A, const _Float16* __restrict__ Wt,
                       const float* __restrict__ bias, float* __restrict__ Cf,
                       _Float16* __restrict__ Ch, int M, int N, int K, int relu,
                       int store_vt) {
  const int wave = threadIdx.x >> 5, lane = threadIdx.x & 31;
  const int hi = lane >> 4, l16 = lane & 15;
  const int tiles_n = N >> 6;                       // 64-wide tiles
  const int tile = blockIdx.x * (TB >> 5) + wave;
  if (tile >= (M >> 4) * tiles_n) return;
  const int m0 = (tile / tiles_n) << 4;
  const int n0 = (tile % tiles_n) << 6;
  const _Float16* arow = A + (size_t)(m0 + l16) * K;

  v8f acc[4] = {{}, {}, {}, {}};
  for (int k0 = 0; k0 < K; k0 += 32) {
    const v16h a = load_frag_h(arow + k0, hi);
#pragma unroll
    for (int nt = 0; nt < 4; ++nt) {
      const v16h b = load_frag_h(Wt + (size_t)(n0 + nt * 16 + l16) * K + k0, hi);
      acc[nt] = WMMA_F16(a, b, acc[nt]);
    }
  }
#pragma unroll
  for (int nt = 0; nt < 4; ++nt) {
    const int n = n0 + nt * 16 + l16;
    const float bv = bias ? bias[n] : 0.0f;
    float vv[8];
#pragma unroll
    for (int r = 0; r < 8; ++r) {
      float v = acc[nt][r] + bv;
      if (relu) v = fmaxf(v, 0.0f);
      vv[r] = v;
    }
    if (Cf) {
#pragma unroll
      for (int r = 0; r < 8; ++r) Cf[(size_t)(m0 + r + (hi << 3)) * N + n] = vv[r];
    }
    if (Ch) {
      if (store_vt) {                               // V: [B,S,(h,d)] -> [B,H,d,S]
        const int b = m0 >> 10, h = n >> 6, d = n & 63;
        const size_t base = ((size_t)((b * N_HEAD + h) * D_Kh + d)) << 10;
#pragma unroll
        for (int r = 0; r < 8; ++r)
          Ch[base + (m0 & 1023) + r + (hi << 3)] = (_Float16)vv[r];
      } else {
#pragma unroll
        for (int r = 0; r < 8; ++r)
          Ch[(size_t)(m0 + r + (hi << 3)) * N + n] = (_Float16)vv[r];
      }
    }
  }
}

// ---- scores = (Q Kᵀ)/8, masked; wave = 16q x 64k; Q,K f16 [B*S, 512] -------
__global__ void attn_scores_h(const _Float16* __restrict__ Q, const _Float16* __restrict__ Kc,
                              const int* __restrict__ mask, float* __restrict__ attn) {
  const int wave = threadIdx.x >> 5, lane = threadIdx.x & 31;
  const int hi = lane >> 4, l16 = lane & 15;
  int id = blockIdx.x * (TB >> 5) + wave;           // 4*8*64*16 waves
  const int kt = id & 15; id >>= 4;
  const int qt = id & 63; id >>= 6;
  const int h  = id & 7;  id >>= 3;
  const int b  = id;
  const int q0 = qt << 4;
  const _Float16* qrow = Q + (size_t)(b * S_LEN + q0 + l16) * D_MODEL + h * D_Kh;
  const v16h a0 = load_frag_h(qrow, hi);            // d = 0..31
  const v16h a1 = load_frag_h(qrow + 32, hi);       // d = 32..63

#pragma unroll
  for (int t = 0; t < 4; ++t) {
    const int k0 = (kt << 6) + t * 16;
    const _Float16* krow = Kc + (size_t)(b * S_LEN + k0 + l16) * D_MODEL + h * D_Kh;
    v8f acc = {};
    acc = WMMA_F16(a0, load_frag_h(krow, hi), acc);
    acc = WMMA_F16(a1, load_frag_h(krow + 32, hi), acc);
#pragma unroll
    for (int r = 0; r < 8; ++r) {
      const int q = q0 + r + (hi << 3), k = k0 + l16;
      float s = acc[r] * 0.125f;                    // 1/sqrt(64)
      if (mask[(size_t)(b * S_LEN + q) * S_LEN + k] == 0) s = -1e9f;
      attn[((size_t)(b * N_HEAD + h) * S_LEN + q) * S_LEN + k] = s;
    }
  }
}

// ---- row softmax then attn = 0.8*softmax + 0.2*extra (wave per row) --------
// Lane owns 8 contiguous float4 chunks -> all accesses are b128.
__global__ void softmax_blend(float* __restrict__ attn, const float* __restrict__ extra) {
  const int wave = threadIdx.x >> 5, lane = threadIdx.x & 31;
  const int row = blockIdx.x * (TB >> 5) + wave;    // (b*8+h)*1024+q
  const int b = row >> 13, q = row & 1023;
  float* arow = attn + (size_t)row * S_LEN;
  const float* erow = extra + (size_t)(b * S_LEN + q) * S_LEN;

  f4 x[8];
  float mx = -INFINITY;
#pragma unroll
  for (int c = 0; c < 8; ++c) {
    x[c] = *(const f4*)(arow + c * 128 + lane * 4);
#pragma unroll
    for (int j = 0; j < 4; ++j) mx = fmaxf(mx, x[c][j]);
  }
  mx = wave_max(mx);
  float s = 0.0f;
#pragma unroll
  for (int c = 0; c < 8; ++c)
#pragma unroll
    for (int j = 0; j < 4; ++j) { x[c][j] = __expf(x[c][j] - mx); s += x[c][j]; }
  const float inv = (1.0f - RATIO_F) / wave_sum(s);
#pragma unroll
  for (int c = 0; c < 8; ++c) {
    const f4 e = *(const f4*)(erow + c * 128 + lane * 4);
    f4 o;
#pragma unroll
    for (int j = 0; j < 4; ++j) o[j] = x[c][j] * inv + RATIO_F * e[j];
    *(f4*)(arow + c * 128 + lane * 4) = o;
  }
}

// ---- ctx[b,q,(h,:)] = attn[b,h,q,:] @ V; Vt f16 [B,H,64,S]; wave = 16x64 ---
__global__ void attn_ctx_h(const float* __restrict__ attn, const _Float16* __restrict__ Vt,
                           _Float16* __restrict__ ctxh) {
  const int wave = threadIdx.x >> 5, lane = threadIdx.x & 31;
  const int hi = lane >> 4, l16 = lane & 15;
  int id = blockIdx.x * (TB >> 5) + wave;           // 4*8*64 waves
  const int qt = id & 63; id >>= 6;
  const int h  = id & 7;  id >>= 3;
  const int b  = id;
  const int q0 = qt << 4;
  const float* arow = attn + ((size_t)(b * N_HEAD + h) * S_LEN + q0 + l16) * S_LEN;
  const _Float16* vbase = Vt + ((size_t)(b * N_HEAD + h) * D_Kh) * S_LEN;

  v8f acc[4] = {{}, {}, {}, {}};
  for (int kk0 = 0; kk0 < S_LEN; kk0 += 32) {
    const v16h a = load_frag_f(arow + kk0, hi);     // probs in [0,1] -> f16 safe
#pragma unroll
    for (int nt = 0; nt < 4; ++nt) {                // B row = output dim d
      const v16h bm = load_frag_h(vbase + (size_t)(nt * 16 + l16) * S_LEN + kk0, hi);
      acc[nt] = WMMA_F16(a, bm, acc[nt]);
    }
  }
#pragma unroll
  for (int nt = 0; nt < 4; ++nt)
#pragma unroll
    for (int r = 0; r < 8; ++r)
      ctxh[(size_t)(b * S_LEN + q0 + r + (hi << 3)) * D_MODEL + h * D_Kh + nt * 16 + l16] =
          (_Float16)acc[nt][r];
}

// ---- out = LayerNorm(a + res)*g + b; optional f16 copy (wave per row) ------
// Lane owns 4 contiguous float4 chunks -> b128 loads/stores, h4 f16 stores.
__global__ void add_layernorm(const float* __restrict__ a, const float* __restrict__ res,
                              const float* __restrict__ g, const float* __restrict__ bta,
                              float* __restrict__ out, _Float16* __restrict__ outh) {
  const int wave = threadIdx.x >> 5, lane = threadIdx.x & 31;
  const int row = blockIdx.x * (TB >> 5) + wave;    // 0..4095
  const size_t base = (size_t)row * D_MODEL;
  f4 x[4];
  float s = 0.0f;
#pragma unroll
  for (int c = 0; c < 4; ++c) {
    const int off = c * 128 + lane * 4;
    const f4 av = *(const f4*)(a + base + off);
    const f4 rv = *(const f4*)(res + base + off);
#pragma unroll
    for (int j = 0; j < 4; ++j) { x[c][j] = av[j] + rv[j]; s += x[c][j]; }
  }
  const float mu = wave_sum(s) * (1.0f / D_MODEL);
  float vs = 0.0f;
#pragma unroll
  for (int c = 0; c < 4; ++c)
#pragma unroll
    for (int j = 0; j < 4; ++j) { const float d = x[c][j] - mu; vs += d * d; }
  const float rstd = rsqrtf(wave_sum(vs) * (1.0f / D_MODEL) + 1e-6f);
  f4 y[4];
#pragma unroll
  for (int c = 0; c < 4; ++c) {
    const int off = c * 128 + lane * 4;
    const f4 gv = *(const f4*)(g + off);
    const f4 bv = *(const f4*)(bta + off);
#pragma unroll
    for (int j = 0; j < 4; ++j) y[c][j] = (x[c][j] - mu) * rstd * gv[j] + bv[j];
    *(f4*)(out + base + off) = y[c];
  }
  if (outh) {
#pragma unroll
    for (int c = 0; c < 4; ++c) {
      h4 o;
#pragma unroll
      for (int j = 0; j < 4; ++j) o[j] = (_Float16)y[c][j];
      *(h4*)(outh + base + c * 128 + lane * 4) = o;
    }
  }
}

// ---------------------------------------------------------------------------
struct AttnP { const float *fc, *fc_b, *ln_b, *ln_g, *wk, *wq, *wv; };

extern "C" void kernel_launch(void* const* d_in, const int* in_sizes, int n_in,
                              void* d_out, int out_size, void* d_ws, size_t ws_size,
                              hipStream_t stream) {
  const float* enc  = (const float*)d_in[0];
  const int*   mask = (const int*)  d_in[1];
  const float* em1  = (const float*)d_in[2];
  const float* em2  = (const float*)d_in[3];
  const float* em3  = (const float*)d_in[4];
  // params pytree (dict keys sorted): attn1{fc,fc_b,ln_b,ln_g,wk,wq,wv}, attn2, attn3,
  // then ffn{b1,b2,ln_b,ln_g,w1,w2}
  auto P = [&](int i) { return (const float*)d_in[i]; };
  AttnP ap[3] = {{P(5),  P(6),  P(7),  P(8),  P(9),  P(10), P(11)},
                 {P(12), P(13), P(14), P(15), P(16), P(17), P(18)},
                 {P(19), P(20), P(21), P(22), P(23), P(24), P(25)}};
  const float *fb1 = P(26), *fb2 = P(27), *fln_b = P(28), *fln_g = P(29),
              *fw1 = P(30), *fw2 = P(31);

  // ---- workspace carve-up (f16 region first, then f32) ----
  const size_t MD = (size_t)M_TOT * D_MODEL;        // 2,097,152
  _Float16* hc = (_Float16*)d_ws;
  auto aH = [&](size_t n) { _Float16* p = hc; hc += n; return p; };
  _Float16 *wq_t[3], *wk_t[3], *wv_t[3], *fc_t[3];
  for (int i = 0; i < 3; ++i) {
    wq_t[i] = aH(MD / 8);  wk_t[i] = aH(MD / 8);    // 512x512 = 262144
    wv_t[i] = aH(MD / 8);  fc_t[i] = aH(MD / 8);
  }
  _Float16* w1_t = aH((size_t)D_INNER * D_MODEL);   // [2048,512]
  _Float16* w2_t = aH((size_t)D_MODEL * D_INNER);   // [512,2048]
  _Float16* ench  = aH(MD);
  _Float16* qh    = aH(MD);
  _Float16* kh    = aH(MD);
  _Float16* vt    = aH(MD);                         // [B,H,64,S]
  _Float16* ctxh  = aH(MD);
  _Float16* out1h = aH(MD);
  _Float16* out2h = aH(MD);
  _Float16* ffhh  = aH((size_t)M_TOT * D_INNER);
  float* fcp = (float*)hc;
  auto aF = [&](size_t n) { float* p = fcp; fcp += n; return p; };
  float* hidb = aF(MD);
  float* out1 = aF(MD);
  float* out2 = aF(MD);
  float* ffh2 = aF(MD);

  float* outO = (float*)d_out;                      // [4,1024,512]
  float* outA = outO + MD;                          // [4,8,1024,1024]
  float* outH = outA + (size_t)4 * N_HEAD * S_LEN * S_LEN;

  // ---- one-time conversions ----
  cvt_f16<<<(int)(MD / 8 / TB), TB, 0, stream>>>(enc, ench, (int)(MD / 8));
  for (int i = 0; i < 3; ++i) {
    wt_cvt<<<(D_MODEL * D_MODEL) / TB, TB, 0, stream>>>(ap[i].wq, wq_t[i], D_MODEL, D_MODEL);
    wt_cvt<<<(D_MODEL * D_MODEL) / TB, TB, 0, stream>>>(ap[i].wk, wk_t[i], D_MODEL, D_MODEL);
    wt_cvt<<<(D_MODEL * D_MODEL) / TB, TB, 0, stream>>>(ap[i].wv, wv_t[i], D_MODEL, D_MODEL);
    wt_cvt<<<(D_MODEL * D_MODEL) / TB, TB, 0, stream>>>(ap[i].fc, fc_t[i], D_MODEL, D_MODEL);
  }
  wt_cvt<<<(D_MODEL * D_INNER) / TB, TB, 0, stream>>>(fw1, w1_t, D_MODEL, D_INNER);
  wt_cvt<<<(D_MODEL * D_INNER) / TB, TB, 0, stream>>>(fw2, w2_t, D_INNER, D_MODEL);

  const int gG512  = (M_TOT / 16) * (D_MODEL / 64) / 8;   // 256 blocks
  const int gG2048 = (M_TOT / 16) * (D_INNER / 64) / 8;   // 1024
  const int gSc    = 4 * N_HEAD * 64 * 16 / 8;            // 4096
  const int gSoft  = 4 * N_HEAD * S_LEN / 8;              // 4096
  const int gCtx   = 4 * N_HEAD * 64 / 8;                 // 256
  const int gLN    = M_TOT / 8;                           // 512

  auto run_mha = [&](const _Float16* qah, const _Float16* kvh, const float* resid,
                     const float* extra, int pi, float* attn_buf, float* hidden_buf,
                     float* out_buf, _Float16* out_bufh) {
    gemm_h<<<gG512, TB, 0, stream>>>(qah, wq_t[pi], nullptr, nullptr, qh, M_TOT, D_MODEL, D_MODEL, 0, 0);
    gemm_h<<<gG512, TB, 0, stream>>>(kvh, wk_t[pi], nullptr, nullptr, kh, M_TOT, D_MODEL, D_MODEL, 0, 0);
    gemm_h<<<gG512, TB, 0, stream>>>(kvh, wv_t[pi], nullptr, nullptr, vt, M_TOT, D_MODEL, D_MODEL, 0, 1);
    attn_scores_h<<<gSc,   TB, 0, stream>>>(qh, kh, mask, attn_buf);
    softmax_blend<<<gSoft, TB, 0, stream>>>(attn_buf, extra);
    attn_ctx_h   <<<gCtx,  TB, 0, stream>>>(attn_buf, vt, ctxh);
    gemm_h<<<gG512, TB, 0, stream>>>(ctxh, fc_t[pi], ap[pi].fc_b, hidden_buf, nullptr,
                                     M_TOT, D_MODEL, D_MODEL, 0, 0);
    add_layernorm<<<gLN, TB, 0, stream>>>(hidden_buf, resid, ap[pi].ln_g, ap[pi].ln_b,
                                          out_buf, out_bufh);
  };

  // attn2 / attn3 over enc_input (d_out attn region used as scores scratch)
  run_mha(ench, ench, enc, em2, 1, outA, hidb, out1, out1h);
  run_mha(ench, ench, enc, em3, 2, outA, hidb, out2, out2h);
  // attn1: q from out2, k/v from out1; attn+hidden are final outputs.
  // out1/out1h are dead after the K/V projections -> reuse as FFN input x.
  run_mha(out2h, out1h, out2, em1, 0, outA, outH, out1, out1h);

  // FFN: out = LN(relu(x@w1+b1)@w2 + b2 + x)
  const float* x = out1;
  gemm_h<<<gG2048, TB, 0, stream>>>(out1h, w1_t, fb1, nullptr, ffhh, M_TOT, D_INNER, D_MODEL, 1, 0);
  gemm_h<<<gG512,  TB, 0, stream>>>(ffhh,  w2_t, fb2, ffh2, nullptr, M_TOT, D_MODEL, D_INNER, 0, 0);
  add_layernorm<<<gLN, TB, 0, stream>>>(ffh2, x, fln_g, fln_b, outO, nullptr);
}
